// MultiHeadAttention_19662360281202
// MI455X (gfx1250) — matrix operand order repaired and checked by reading, not compiled, not run on hardware
//
#include <hip/hip_runtime.h>
#include <hip/hip_bf16.h>
#include <stdint.h>

// ---------------------------------------------------------------------------
// Types for CDNA5 WMMA (wave32): 16x16x32 bf16 -> f32 accumulate
// ---------------------------------------------------------------------------
typedef __attribute__((ext_vector_type(16))) __bf16 v16bf;
typedef __attribute__((ext_vector_type(8)))  float  v8f;
typedef __attribute__((ext_vector_type(4)))  int    v4i;

union Frag {
    v16bf v;
    uint4 u[2];   // two 16B halves for aligned LDS/global b128 loads
};

__device__ __forceinline__ uint16_t f2b(float f) {
    uint32_t u = __builtin_bit_cast(uint32_t, f);
    u += 0x7FFFu + ((u >> 16) & 1u);          // round-to-nearest-even
    return (uint16_t)(u >> 16);
}
__device__ __forceinline__ float b2f(uint16_t b) {
    uint32_t u = ((uint32_t)b) << 16;
    return __builtin_bit_cast(float, u);
}

// ---------------------------------------------------------------------------
// CDNA5 direct global->LDS async copy (ASYNCcnt path), with safe fallback.
// Builtin signature (from probe diagnostics): (v4i AS1*, v4i AS3*, imm, imm).
// ---------------------------------------------------------------------------
#if defined(__has_builtin)
#  if __has_builtin(__builtin_amdgcn_global_load_async_to_lds_b128)
#    define ASYNC_COPY 1
#  endif
#endif
#ifndef ASYNC_COPY
#  define ASYNC_COPY 0
#endif

#if ASYNC_COPY
typedef __attribute__((address_space(1))) v4i glb_v4i;
typedef __attribute__((address_space(3))) v4i lds_v4i;

__device__ __forceinline__ void async_cp16(const uint16_t* g, uint16_t* l) {
    // generic -> addrspace(1)/(3) via integer round-trip; LDS offset is the
    // low 32 bits of the generic pointer (flat aperture rule).
    __builtin_amdgcn_global_load_async_to_lds_b128(
        (glb_v4i*)(uintptr_t)g,
        (lds_v4i*)(uintptr_t)l,
        0, 0);
}
#  if __has_builtin(__builtin_amdgcn_s_wait_asynccnt)
#    define WAIT_ASYNC() __builtin_amdgcn_s_wait_asynccnt(0)
#  else
#    define WAIT_ASYNC() asm volatile("s_wait_asynccnt 0" ::: "memory")
#  endif
#endif

// ---------------------------------------------------------------------------
// fp32 -> bf16 cast, 4 elements/thread
// ---------------------------------------------------------------------------
__global__ void cast_f32_to_bf16(const float* __restrict__ in,
                                 uint16_t* __restrict__ out, int n4) {
    int i = blockIdx.x * blockDim.x + threadIdx.x;
    if (i >= n4) return;
    float4 f = ((const float4*)in)[i];
    ushort4 o;
    o.x = f2b(f.x); o.y = f2b(f.y); o.z = f2b(f.z); o.w = f2b(f.w);
    ((ushort4*)out)[i] = o;
}

// ---------------------------------------------------------------------------
// C[m,n] = sum_k A[m,k]*B[n,k] + bias[n]
// A: MxK bf16 row-major, B: NxK bf16 row-major (i.e. computes A @ B^T)
// Block tile 128x128, BK=32, 8 waves x (32x64) sub-tiles, double-buffered LDS
// filled via GLOBAL_LOAD_ASYNC_TO_LDS_B128 (ASYNCcnt) when available.
// ---------------------------------------------------------------------------
#define BM 128
#define BN 128
#define BK 32
#define LDT 40   // padded LDS K-stride (elements) to spread banks

template<int OUT_BF16>
__global__ __launch_bounds__(256)
void gemm_bt(const uint16_t* __restrict__ A, const uint16_t* __restrict__ B,
             const float* __restrict__ bias, void* __restrict__ Cout,
             int M, int N, int K)
{
    __shared__ __align__(16) uint16_t sA[2][BM * LDT];
    __shared__ __align__(16) uint16_t sB[2][BN * LDT];

    const int t    = threadIdx.x;
    const int lane = t & 31;
    const int w    = t >> 5;                // wave 0..7
    const int m0   = blockIdx.y * BM;
    const int n0   = blockIdx.x * BN;

    // global->LDS staging: 256 threads, each moves 2x16B for A and B per stage
    const int ldrow = t >> 2;               // 0..63
    const int ldk   = (t & 3) * 8;          // 0,8,16,24 (bf16 elements)
    const uint16_t* gA = A + (size_t)(m0 + ldrow) * K + ldk;
    const uint16_t* gB = B + (size_t)(n0 + ldrow) * K + ldk;
    const size_t rowStep = (size_t)64 * K;

    uint16_t* lA0 = &sA[0][ldrow * LDT + ldk];
    uint16_t* lA1 = &sA[0][(ldrow + 64) * LDT + ldk];
    uint16_t* lB0 = &sB[0][ldrow * LDT + ldk];
    uint16_t* lB1 = &sB[0][(ldrow + 64) * LDT + ldk];
    const int bufStrideA = BM * LDT;        // elements between buffer 0 and 1
    const int bufStrideB = BN * LDT;

    v8f acc[2][4] = {};

    // ---- prologue: stage 0 -> buffer 0 ----
#if ASYNC_COPY
    async_cp16(gA,           lA0);
    async_cp16(gA + rowStep, lA1);
    async_cp16(gB,           lB0);
    async_cp16(gB + rowStep, lB1);
    WAIT_ASYNC();
#else
    *(uint4*)lA0 = *(const uint4*)(gA);
    *(uint4*)lA1 = *(const uint4*)(gA + rowStep);
    *(uint4*)lB0 = *(const uint4*)(gB);
    *(uint4*)lB1 = *(const uint4*)(gB + rowStep);
#endif
    __syncthreads();

    // per-wave fragment coordinates (16-bit WMMA VGPR layouts)
    const int mrow  = (w >> 1) * 32 + (lane & 15);  // A row within block tile
    const int ncol  = (w & 1) * 64 + (lane & 15);   // B row (output column)
    const int koffA = (lane >> 4) * 8;              // A: K split 0..7 / 8..15
    const int koffB = (lane >> 4) * 8;              // B: K split 0..7 / 8..15

    int buf = 0;
    for (int ks = 0; ks < K; ks += BK) {
        const bool more = (ks + BK) < K;
        const int nb = buf ^ 1;

#if ASYNC_COPY
        // kick next stage into the back buffer; overlaps with WMMA below
        if (more) {
            const int kn = ks + BK;
            async_cp16(gA + kn,           lA0 + nb * bufStrideA);
            async_cp16(gA + kn + rowStep, lA1 + nb * bufStrideA);
            async_cp16(gB + kn,           lB0 + nb * bufStrideB);
            async_cp16(gB + kn + rowStep, lB1 + nb * bufStrideB);
        }
#else
        uint4 ra0{}, ra1{}, rb0{}, rb1{};
        if (more) {
            const uint16_t* pA = gA + (ks + BK);
            const uint16_t* pB = gB + (ks + BK);
            ra0 = *(const uint4*)(pA);
            ra1 = *(const uint4*)(pA + rowStep);
            rb0 = *(const uint4*)(pB);
            rb1 = *(const uint4*)(pB + rowStep);
        }
#endif

        // ---- compute from front buffer ----
        Frag a[2], b[4];
#pragma unroll
        for (int i = 0; i < 2; ++i) {
            const uint16_t* p = &sA[buf][(mrow + i * 16) * LDT + koffA];
            a[i].u[0] = *(const uint4*)(p);
            a[i].u[1] = *(const uint4*)(p + 16);   // K += 16
        }
#pragma unroll
        for (int j = 0; j < 4; ++j) {
            const uint16_t* p = &sB[buf][(ncol + j * 16) * LDT + koffB];
            b[j].u[0] = *(const uint4*)(p);
            b[j].u[1] = *(const uint4*)(p + 16);   // K += 16
        }
#pragma unroll
        for (int i = 0; i < 2; ++i)
#pragma unroll
            for (int j = 0; j < 4; ++j)
                acc[i][j] = __builtin_amdgcn_wmma_f32_16x16x32_bf16(
                    false, a[i].v, false, b[j].v, (short)0, acc[i][j],
                    false, false);

#if ASYNC_COPY
        if (more) WAIT_ASYNC();
#else
        if (more) {
            *(uint4*)(lA0 + nb * bufStrideA) = ra0;
            *(uint4*)(lA1 + nb * bufStrideA) = ra1;
            *(uint4*)(lB0 + nb * bufStrideB) = rb0;
            *(uint4*)(lB1 + nb * bufStrideB) = rb1;
        }
#endif
        __syncthreads();
        buf = nb;
    }

    // epilogue: C/D layout -> lane holds column (lane&15), rows (lane>>4)*8+v
    const int erow0 = m0 + (w >> 1) * 32 + (lane >> 4) * 8;
#pragma unroll
    for (int i = 0; i < 2; ++i) {
#pragma unroll
        for (int j = 0; j < 4; ++j) {
            const int col = n0 + (w & 1) * 64 + j * 16 + (lane & 15);
            const float bv = bias[col];
#pragma unroll
            for (int v = 0; v < 8; ++v) {
                const int row = erow0 + i * 16 + v;
                const float val = acc[i][j][v] + bv;
                if (OUT_BF16)
                    ((uint16_t*)Cout)[(size_t)row * N + col] = f2b(val);
                else
                    ((float*)Cout)[(size_t)row * N + col] = val;
            }
        }
    }
}

// ---------------------------------------------------------------------------
// Per-(batch, seq) attention over HEADS (the reference's buggy einsum):
//   scores[q,k] = (1/sqrt(128)) * sum_d Q[n,q,s,d]*K[n,k,s,d]   (16x16)
//   p = softmax_k(scores);  ctx[q,d] = sum_k p[q,k]*V[n,k,s,d]  (16x128)
// ctx is stored bf16 in the scrambled reshape order so the output projection
// is a plain row-major GEMM:
//   row = n*2048 + q*128 + s/16 ; col = (s%16)*128 + d
// One wave (32 lanes) per (n,s); scores via 4 chained bf16 WMMAs.
// ---------------------------------------------------------------------------
__global__ __launch_bounds__(32)
void attn_heads(const uint16_t* __restrict__ Q, const uint16_t* __restrict__ Km,
                const uint16_t* __restrict__ V, uint16_t* __restrict__ ctx)
{
    __shared__ float sS[16][17];
    const int lane = threadIdx.x;
    const int bid  = blockIdx.x;
    const int n    = bid >> 11;       // batch
    const int s    = bid & 2047;      // seq position
    const size_t rbase = ((size_t)n * 2048 + s) * 2048;  // row in (B*S, E)

    // scores = Q_heads(16x128) @ K_heads^T(128x16)
    v8f acc = {};
    const int qrow  = lane & 15;           // A row = q head, B col = k head
    const int koffA = (lane >> 4) * 8;
    const int koffB = (lane >> 4) * 8;
#pragma unroll
    for (int st = 0; st < 4; ++st) {
        const int kb = st * 32;
        Frag a, b;
        const uint16_t* pa = Q  + rbase + qrow * 128 + kb + koffA;
        a.u[0] = *(const uint4*)(pa);
        a.u[1] = *(const uint4*)(pa + 16);
        const uint16_t* pb = Km + rbase + qrow * 128 + kb + koffB;
        b.u[0] = *(const uint4*)(pb);
        b.u[1] = *(const uint4*)(pb + 16);
        acc = __builtin_amdgcn_wmma_f32_16x16x32_bf16(
            false, a.v, false, b.v, (short)0, acc, false, false);
    }
    const float scale = 0.08838834764831845f;  // 1/sqrt(128)
#pragma unroll
    for (int v = 0; v < 8; ++v)
        sS[(lane >> 4) * 8 + v][lane & 15] = acc[v] * scale;
    __syncthreads();

    // softmax over k (16 heads), one row per lane 0..15
    if (lane < 16) {
        float m = -3.4e38f;
#pragma unroll
        for (int k = 0; k < 16; ++k) m = fmaxf(m, sS[lane][k]);
        float e[16];
        float sum = 0.f;
#pragma unroll
        for (int k = 0; k < 16; ++k) { e[k] = __expf(sS[lane][k] - m); sum += e[k]; }
        const float inv = 1.f / sum;
#pragma unroll
        for (int k = 0; k < 16; ++k) sS[lane][k] = e[k] * inv;
    }
    __syncthreads();

    // ctx: lane covers d = lane + 32*j, j=0..3
    float vv[16][4];
#pragma unroll
    for (int k = 0; k < 16; ++k)
#pragma unroll
        for (int j = 0; j < 4; ++j)
            vv[k][j] = b2f(V[rbase + k * 128 + lane + 32 * j]);

    const int orow0 = n * 2048 + (s >> 4);          // + q*128
    const int ocol0 = (s & 15) * 128 + lane;        // + 32*j
#pragma unroll
    for (int q = 0; q < 16; ++q) {
        float f0 = 0.f, f1 = 0.f, f2 = 0.f, f3 = 0.f;
#pragma unroll
        for (int k = 0; k < 16; ++k) {
            const float p = sS[q][k];
            f0 += p * vv[k][0]; f1 += p * vv[k][1];
            f2 += p * vv[k][2]; f3 += p * vv[k][3];
        }
        uint16_t* o = ctx + (size_t)(orow0 + q * 128) * 2048 + ocol0;
        o[0]  = f2b(f0);
        o[32] = f2b(f1);
        o[64] = f2b(f2);
        o[96] = f2b(f3);
    }
}

// ---------------------------------------------------------------------------
// Host launcher
// ---------------------------------------------------------------------------
extern "C" void kernel_launch(void* const* d_in, const int* in_sizes, int n_in,
                              void* d_out, int out_size, void* d_ws, size_t ws_size,
                              hipStream_t stream)
{
    (void)in_sizes; (void)n_in; (void)out_size; (void)ws_size;

    const float* x  = (const float*)d_in[0];
    const float* Wq = (const float*)d_in[1];
    const float* bq = (const float*)d_in[2];
    const float* Wk = (const float*)d_in[3];
    const float* bk = (const float*)d_in[4];
    const float* Wv = (const float*)d_in[5];
    const float* bv = (const float*)d_in[6];
    const float* Wo = (const float*)d_in[7];
    const float* bo = (const float*)d_in[8];
    float* out = (float*)d_out;

    const int M = 4 * 2048;                 // batch * seq rows
    const int E = 2048;
    const size_t SZX = (size_t)M * E;       // activation elems
    const size_t SZW = (size_t)E * E;       // weight elems

    // workspace layout (bf16 elements): 192 MB total
    uint16_t* xb  = (uint16_t*)d_ws;
    uint16_t* Wqb = xb  + SZX;
    uint16_t* Wkb = Wqb + SZW;
    uint16_t* Wvb = Wkb + SZW;
    uint16_t* Wob = Wvb + SZW;
    uint16_t* Qb  = Wob + SZW;
    uint16_t* Kb  = Qb  + SZX;
    uint16_t* Vb  = Kb  + SZX;
    uint16_t* Cb  = Vb  + SZX;

    // 1) fp32 -> bf16 casts
    {
        int n4 = (int)(SZX / 4);
        cast_f32_to_bf16<<<(n4 + 255) / 256, 256, 0, stream>>>(x, xb, n4);
        n4 = (int)(SZW / 4);
        cast_f32_to_bf16<<<(n4 + 255) / 256, 256, 0, stream>>>(Wq, Wqb, n4);
        cast_f32_to_bf16<<<(n4 + 255) / 256, 256, 0, stream>>>(Wk, Wkb, n4);
        cast_f32_to_bf16<<<(n4 + 255) / 256, 256, 0, stream>>>(Wv, Wvb, n4);
        cast_f32_to_bf16<<<(n4 + 255) / 256, 256, 0, stream>>>(Wo, Wob, n4);
    }

    // 2) Q/K/V projections (bf16 WMMA GEMMs, bf16 output)
    dim3 gg(E / BN, M / BM);
    gemm_bt<1><<<gg, 256, 0, stream>>>(xb, Wqb, bq, Qb, M, E, E);
    gemm_bt<1><<<gg, 256, 0, stream>>>(xb, Wkb, bk, Kb, M, E, E);
    gemm_bt<1><<<gg, 256, 0, stream>>>(xb, Wvb, bv, Vb, M, E, E);

    // 3) per-position head attention + scrambled-reshape store
    attn_heads<<<M, 32, 0, stream>>>(Qb, Kb, Vb, Cb);

    // 4) output projection (fp32 output to d_out)
    gemm_bt<0><<<gg, 256, 0, stream>>>(Cb, Wob, bo, out, M, E, E);
}
